// CLIPDenseBase_39771397161213
// MI455X (gfx1250) — matrix operand order repaired and checked by reading, not compiled, not run on hardware
//
#include <hip/hip_runtime.h>
#include <math.h>

typedef unsigned short u16;
typedef __attribute__((ext_vector_type(16))) __bf16 v16bf;
typedef __attribute__((ext_vector_type(8)))  float  v8f;
typedef __attribute__((ext_vector_type(4)))  unsigned int v4u;
typedef __attribute__((ext_vector_type(8)))  int v8i;
typedef __attribute__((ext_vector_type(4)))  int v4i;

#define DD     768
#define BB     64
#define HH     12
#define NTOK   197
#define TOK    (BB * NTOK)     /* 12608 */
#define NPATCH 196
#define PTOK   (BB * NPATCH)   /* 12544 */
#define TD     2304
#define MLPD   3072
#define NL     12
#define SLD    224             /* padded score row length (floats); also P row length (bf16) */
#define PLD    448             /* P row stride in u16 (== SLD floats) */

#define LROW   40              /* LDS tile row pitch in u16: 32 data + 4-DWORD TDM pad */
#define TILE_HALF (128 * LROW) /* one 128x32 tile in LDS, u16 units = 5120 */

#if __has_builtin(__builtin_amdgcn_tensor_load_to_lds)
#define HAS_TDM 1
#else
#define HAS_TDM 0
#endif

__device__ __forceinline__ u16 f2bf(float f) {
    unsigned u = __float_as_uint(f);
    u += 0x7FFFu + ((u >> 16) & 1u);   // round-to-nearest-even
    return (u16)(u >> 16);
}
__device__ __forceinline__ int imin(int a, int b) { return a < b ? a : b; }

union BfTile {
    uint4 q[2];
    u16   u[16];
    v16bf v;
};

#if HAS_TDM
// Issue one TDM 2D tile load: tile_dim0=32 (K), tile_dim1=128 (rows), 2B elems,
// zero-fill rows >= dim1, LDS pad 4 DWORDs per 16 DWORDs (row pitch 80B).
__device__ __forceinline__ void tdm_load_tile(unsigned lds_off, const u16* gptr,
                                              unsigned dim1, unsigned long long stride0)
{
    const unsigned long long ga = (unsigned long long)(size_t)gptr;
    v4u g0;
    g0[0] = 1u;                                    // count=1
    g0[1] = lds_off;                               // LDS byte address
    g0[2] = (unsigned)ga;                          // global addr [31:0]
    g0[3] = (unsigned)(ga >> 32) | (2u << 30);     // global addr [56:32] | type=2
    v8i g1;
    g1[0] = (1 << 16) | (1 << 20) | (3 << 22) | (3 << 25); // data_size=2B, pad_en, ivl=16DW, amt=4DW
    g1[1] = (int)(32u << 16);                      // tensor_dim0 = 32 (== tile_dim0, no X OOB)
    g1[2] = (int)((dim1 & 0xFFFFu) << 16);         // tensor_dim1[15:0]
    g1[3] = (int)((dim1 >> 16) | (32u << 16));     // tensor_dim1[31:16] | tile_dim0=32
    g1[4] = 128;                                   // tile_dim1 = 128, tile_dim2 = 0
    g1[5] = (int)(stride0 & 0xFFFFFFFFull);        // tensor_dim0_stride[31:0] (elements)
    g1[6] = (int)((stride0 >> 32) & 0xFFFFull);    // tensor_dim0_stride[47:32]
    g1[7] = 0;
    const v4i gz4 = {0, 0, 0, 0};
    const v8i gz8 = {0, 0, 0, 0, 0, 0, 0, 0};
    __builtin_amdgcn_tensor_load_to_lds(g0, g1, gz4, gz4, gz8, 0);
}
#endif

// ---------------------------------------------------------------------------
// Mode-0 GEMM:  C[M,N] = act( A[M,K] @ W[N,K]^T + bias ) (+ Res)
// bf16 in / f32 acc. WG = 8 waves, tile 128(M) x 128(N); wave tile 32x64.
// A & W tiles staged to LDS by the Tensor Data Mover (double buffered,
// s_wait_tensorcnt + one barrier per K-step); TDM zero-fills OOB rows.
// K must be a multiple of 32.
// ---------------------------------------------------------------------------
__global__ __launch_bounds__(256) void gemm_nt_tdm(
    const u16* __restrict__ A, long lda,
    const u16* __restrict__ B, long ldb,
    const float* __restrict__ bias,
    float* Cf, u16* Cb, long ldc,
    const float* Res,
    int M, int N, int K, int act,
    int zdiv, long sA0, long sA1, long sB0, long sB1, long sC0, long sC1)
{
    __shared__ u16 smem[4 * TILE_HALF];            // {buf0,buf1} x {A,B}: 40 KB

    const int z  = blockIdx.z;
    const int z0 = z / zdiv;
    const int z1 = z - z0 * zdiv;
    A += z0 * sA0 + z1 * sA1;
    B += z0 * sB0 + z1 * sB1;
    const long coff = z0 * sC0 + z1 * sC1;

    const int lane = threadIdx.x;
    const int w    = threadIdx.y;
    const int rowStart = blockIdx.y * 128;
    const int colStart = blockIdx.x * 128;

    const u16* Abase = A + (long)rowStart * lda;
    const u16* Bbase = B + (long)colStart * ldb;
    const unsigned dimA1 = (unsigned)(M - rowStart);
    const unsigned dimB1 = (unsigned)(N - colStart);

    const int r  = lane & 15;
    const int hi = lane >> 4;
    const int ka = 8  * hi;                        // A K-split: 0..7 / 8..15 (+16)
    const int kb = 16 * hi;                        // unused here: B is loaded with ka, same K split as A
    const int wm = w >> 1;                         // 0..3 -> 32-row slice
    const int wn = w & 1;                          // 0..1 -> 64-col slice

    v8f c[2][4] = {};
    const int nsteps = K >> 5;

#if HAS_TDM
    const unsigned ldsBase = (unsigned)(size_t)(void*)smem;
    auto issue = [&](int k0, int buf) {
        tdm_load_tile(ldsBase + (unsigned)(buf * 2 * TILE_HALF) * 2u,             Abase + k0, dimA1, (unsigned long long)lda);
        tdm_load_tile(ldsBase + (unsigned)(buf * 2 * TILE_HALF + TILE_HALF) * 2u, Bbase + k0, dimB1, (unsigned long long)ldb);
    };
    if (w == 0) issue(0, 0);
    for (int s = 0; s < nsteps; ++s) {
        const int buf = s & 1;
        __builtin_amdgcn_s_wait_tensorcnt(0);      // tile(s) in `buf` complete (wave0; NOP elsewhere)
        __syncthreads();
        if (w == 0 && s + 1 < nsteps) issue((s + 1) << 5, buf ^ 1);
        const u16* sa = smem + buf * 2 * TILE_HALF;
        const u16* sb = sa + TILE_HALF;
#else
    // Fallback: cooperative global->LDS staging (single buffered).
    for (int s = 0; s < nsteps; ++s) {
        const int k0 = s << 5;
        __syncthreads();                           // previous compute done before overwrite
        {
            const int t = w * 32 + lane;           // 0..255
            const uint4 z4 = make_uint4(0u, 0u, 0u, 0u);
#pragma unroll
            for (int ii = 0; ii < 2; ++ii) {
                const int sgi = t + ii * 256;      // 0..511: 128 rows x 4 x 16B
                const int row = sgi >> 2, qo = (sgi & 3) * 8;
                uint4 va = (row < (int)dimA1) ? *(const uint4*)(Abase + (long)row * lda + k0 + qo) : z4;
                uint4 vb = (row < (int)dimB1) ? *(const uint4*)(Bbase + (long)row * ldb + k0 + qo) : z4;
                *(uint4*)(smem + row * LROW + qo) = va;
                *(uint4*)(smem + TILE_HALF + row * LROW + qo) = vb;
            }
        }
        __syncthreads();
        const u16* sa = smem;
        const u16* sb = smem + TILE_HALF;
#endif
        BfTile a[2], b[4];
#pragma unroll
        for (int i = 0; i < 2; ++i) {
            const u16* pa = sa + (wm * 32 + 16 * i + r) * LROW;
            a[i].q[0] = *(const uint4*)(pa + ka);
            a[i].q[1] = *(const uint4*)(pa + 16 + ka);
        }
#pragma unroll
        for (int j = 0; j < 4; ++j) {
            const u16* pb = sb + (wn * 64 + 16 * j + r) * LROW + ka;
            b[j].q[0] = *(const uint4*)(pb);
            b[j].q[1] = *(const uint4*)(pb + 16);
        }
#pragma unroll
        for (int i = 0; i < 2; ++i)
#pragma unroll
            for (int j = 0; j < 4; ++j)
                c[i][j] = __builtin_amdgcn_wmma_f32_16x16x32_bf16(
                    false, a[i].v, false, b[j].v, (short)0, c[i][j], false, false);
#if !HAS_TDM
        __syncthreads();
#endif
    }

    float*       CfB  = Cf  ? Cf  + coff : nullptr;
    u16*         CbB  = Cb  ? Cb  + coff : nullptr;
    const float* ResB = Res ? Res + coff : nullptr;

#pragma unroll
    for (int i = 0; i < 2; ++i)
#pragma unroll
        for (int j = 0; j < 4; ++j) {
            const int n = colStart + wn * 64 + 16 * j + r;
            if (n >= N) continue;
            const float bv = bias ? bias[n] : 0.f;
#pragma unroll
            for (int rr = 0; rr < 8; ++rr) {
                const int m = rowStart + wm * 32 + 16 * i + 8 * hi + rr;
                if (m >= M) continue;
                float v = c[i][j][rr] + bv;
                if (act == 1) v = v * (1.f / (1.f + __expf(-1.702f * v)));   // QuickGELU
                const long idx = (long)m * ldc + n;
                if (ResB) v += ResB[idx];
                if (CfB) CfB[idx] = v;
                else     CbB[idx] = f2bf(v);
            }
        }
}

// ---------------------------------------------------------------------------
// Mode-1 GEMM:  C[M,N] = A[M,K] @ V[K,N]   (V row-major over K rows)
// Small GEMMs only (attention P@V, final projection). Direct loads; B walked
// with an incremented pointer; K rows >= Kvalid clamp by halting the advance.
// ---------------------------------------------------------------------------
__global__ __launch_bounds__(256) void gemm_kn(
    const u16* __restrict__ A, long lda,
    const u16* __restrict__ B, long ldb, int Kvalid,
    const float* __restrict__ bias,
    float* Cf, u16* Cb, long ldc,
    int M, int N, int K,
    int zdiv, long sA0, long sA1, long sB0, long sB1, long sC0, long sC1)
{
    const int z  = blockIdx.z;
    const int z0 = z / zdiv;
    const int z1 = z - z0 * zdiv;
    A += z0 * sA0 + z1 * sA1;
    B += z0 * sB0 + z1 * sB1;
    const long coff = z0 * sC0 + z1 * sC1;

    const int lane = threadIdx.x;
    const int w    = threadIdx.y;
    const int mW   = blockIdx.y * 128 + (w >> 1) * 32;
    const int nW   = blockIdx.x * 64  + (w & 1) * 32;

    const int r  = lane & 15;
    const int hi = lane >> 4;
    const int ka = 8  * hi;
    const int kb = 16 * hi;

    const u16* Ar0 = A + (long)imin(mW + r,      M - 1) * lda;
    const u16* Ar1 = A + (long)imin(mW + 16 + r, M - 1) * lda;
    const int  bn0 = imin(nW + r,      N - 1);
    const int  bn1 = imin(nW + 16 + r, N - 1);

    v8f c00 = {}; v8f c01 = {}; v8f c10 = {}; v8f c11 = {};

    const u16* p0 = B + (long)kb * ldb + bn0;      // walked by += ldb (clamped at Kvalid)
    const u16* p1 = B + (long)kb * ldb + bn1;
    int krow = kb;                                  // current B row index of p0/p1

    for (int k0 = 0; k0 < K; k0 += 32) {
        BfTile a0, a1, b0, b1;
        a0.q[0] = *(const uint4*)(Ar0 + k0 + kb);
        a0.q[1] = *(const uint4*)(Ar0 + k0 + 8 + kb);
        a1.q[0] = *(const uint4*)(Ar1 + k0 + kb);
        a1.q[1] = *(const uint4*)(Ar1 + k0 + 8 + kb);
#pragma unroll
        for (int j = 0; j < 16; ++j) {
            b0.u[j] = *p0;
            b1.u[j] = *p1;
            if (krow + 1 < Kvalid) { p0 += ldb; p1 += ldb; krow++; }
        }
        // advance from (k0+kb+16) to (k0+32+kb): +16 rows, clamped
        const int tgt = imin(k0 + 32 + kb, Kvalid - 1);
        p0 += (long)(tgt - krow) * ldb;
        p1 += (long)(tgt - krow) * ldb;
        krow = tgt;
        c00 = __builtin_amdgcn_wmma_f32_16x16x32_bf16(false, a0.v, false, b0.v, (short)0, c00, false, false);
        c01 = __builtin_amdgcn_wmma_f32_16x16x32_bf16(false, a0.v, false, b1.v, (short)0, c01, false, false);
        c10 = __builtin_amdgcn_wmma_f32_16x16x32_bf16(false, a1.v, false, b0.v, (short)0, c10, false, false);
        c11 = __builtin_amdgcn_wmma_f32_16x16x32_bf16(false, a1.v, false, b1.v, (short)0, c11, false, false);
    }

    float* CfB = Cf ? Cf + coff : nullptr;
    u16*   CbB = Cb ? Cb + coff : nullptr;

    auto store_tile = [&](v8f cc, int ti, int tj) {
        const int n = nW + 16 * tj + r;
        if (n >= N) return;
        const float bv = bias ? bias[n] : 0.f;
#pragma unroll
        for (int rr = 0; rr < 8; ++rr) {
            const int m = mW + 16 * ti + 8 * hi + rr;
            if (m >= M) continue;
            float v = cc[rr] + bv;
            const long idx = (long)m * ldc + n;
            if (CfB) CfB[idx] = v;
            else     CbB[idx] = f2bf(v);
        }
    };
    store_tile(c00, 0, 0); store_tile(c01, 0, 1);
    store_tile(c10, 1, 0); store_tile(c11, 1, 1);
}

// ---------------------------------------------------------------------------
__global__ __launch_bounds__(256) void layernorm768(
    const float* X, long ldx,
    const float* __restrict__ g, const float* __restrict__ bta,
    float* Of, u16* Ob, long ldo)
{
    const int row = blockIdx.x;
    const int tid = threadIdx.x;
    const float* xr = X + (long)row * ldx;
    float v0 = xr[tid], v1 = xr[tid + 256], v2 = xr[tid + 512];
    __shared__ float sb[256];
    sb[tid] = v0 + v1 + v2;
    __syncthreads();
    for (int o = 128; o > 0; o >>= 1) { if (tid < o) sb[tid] += sb[tid + o]; __syncthreads(); }
    const float mean = sb[0] * (1.f / 768.f);
    __syncthreads();
    v0 -= mean; v1 -= mean; v2 -= mean;
    sb[tid] = v0 * v0 + v1 * v1 + v2 * v2;
    __syncthreads();
    for (int o = 128; o > 0; o >>= 1) { if (tid < o) sb[tid] += sb[tid + o]; __syncthreads(); }
    const float rstd = rsqrtf(sb[0] * (1.f / 768.f) + 1e-5f);
    const float y0 = v0 * rstd * g[tid]       + bta[tid];
    const float y1 = v1 * rstd * g[tid + 256] + bta[tid + 256];
    const float y2 = v2 * rstd * g[tid + 512] + bta[tid + 512];
    if (Of) { float* o = Of + (long)row * ldo; o[tid] = y0; o[tid + 256] = y1; o[tid + 512] = y2; }
    else    { u16*   o = Ob + (long)row * ldo; o[tid] = f2bf(y0); o[tid + 256] = f2bf(y1); o[tid + 512] = f2bf(y2); }
}

__global__ __launch_bounds__(256) void softmax_mask(float* S, const float* __restrict__ mask)
{
    const int row = blockIdx.x;                    // (b*H + h)*197 + q
    const int tid = threadIdx.x;
    const int q   = row % NTOK;
    const int bh  = row / NTOK;
    const int mb  = bh % BB;                       // (b*H+h) % B
    float* sr = S + (long)row * SLD;
    float s = -3.0e38f;
    if (tid < NTOK) {
        s = sr[tid] * 0.125f;                      // HEAD_DIM^-0.5
        if (q >= 1 && tid >= 1) s *= mask[(long)mb * NPATCH + tid - 1];
    }
    __shared__ float sb[256];
    sb[tid] = s; __syncthreads();
    for (int o = 128; o > 0; o >>= 1) { if (tid < o) sb[tid] = fmaxf(sb[tid], sb[tid + o]); __syncthreads(); }
    const float mx = sb[0];
    __syncthreads();
    const float e = (tid < NTOK) ? __expf(s - mx) : 0.f;
    sb[tid] = e; __syncthreads();
    for (int o = 128; o > 0; o >>= 1) { if (tid < o) sb[tid] += sb[tid + o]; __syncthreads(); }
    const float inv = 1.f / sb[0];
    __syncthreads();
    u16* pr = (u16*)sr + SLD;                      // bf16 probs in floats [112..224) of this row
    if (tid < NTOK)     pr[tid] = f2bf(e * inv);
    else if (tid < SLD) pr[tid] = 0;               // zero pad -> K=224 GEMM exact
}

__global__ __launch_bounds__(256) void f32_to_bf16k(const float* __restrict__ s, u16* __restrict__ d, long n)
{
    const long i = (long)blockIdx.x * 256 + threadIdx.x;
    if (i < n) d[i] = f2bf(s[i]);
}

__global__ __launch_bounds__(256) void im2col_patches(const float* __restrict__ x, u16* __restrict__ P)
{
    const long i = (long)blockIdx.x * 256 + threadIdx.x;
    if (i >= (long)PTOK * DD) return;
    const int  k  = (int)(i % DD);
    const long p  = i / DD;
    const int  gx = (int)(p % 14);
    const long t  = p / 14;
    const int  gy = (int)(t % 14);
    const int  b  = (int)(t / 14);
    const int  px = k & 15, py = (k >> 4) & 15, c = k >> 8;
    P[i] = f2bf(x[(((long)b * 3 + c) * 224 + (gy * 16 + py)) * 224 + gx * 16 + px]);
}

__global__ __launch_bounds__(256) void assemble_embed(
    const float* __restrict__ patch, const float* __restrict__ cls,
    const float* __restrict__ pos, float* __restrict__ X)
{
    const long i = (long)blockIdx.x * 256 + threadIdx.x;
    if (i >= (long)TOK * DD) return;
    const int  d   = (int)(i % DD);
    const long t   = i / DD;
    const int  tok = (int)(t % NTOK);
    const long b   = t / NTOK;
    const float v  = (tok == 0) ? cls[d] : patch[((long)b * NPATCH + tok - 1) * DD + d];
    X[i] = v + pos[(long)tok * DD + d];
}

extern "C" void kernel_launch(void* const* d_in, const int* in_sizes, int n_in,
                              void* d_out, int out_size, void* d_ws, size_t ws_size,
                              hipStream_t stream)
{
    const float* x_inp   = (const float*)d_in[0];
    const float* maskp   = (const float*)d_in[1];
    const float* conv_w  = (const float*)d_in[2];
    const float* cls     = (const float*)d_in[3];
    const float* pos     = (const float*)d_in[4];
    const float* lnpre_g = (const float*)d_in[5];
    const float* lnpre_b = (const float*)d_in[6];
    const float* ln1_g   = (const float*)d_in[7];
    const float* ln1_b   = (const float*)d_in[8];
    const float* w_in    = (const float*)d_in[9];
    const float* b_in    = (const float*)d_in[10];
    const float* w_out   = (const float*)d_in[11];
    const float* b_out   = (const float*)d_in[12];
    const float* ln2_g   = (const float*)d_in[13];
    const float* ln2_b   = (const float*)d_in[14];
    const float* w_fc    = (const float*)d_in[15];
    const float* b_fc    = (const float*)d_in[16];
    const float* w_cp    = (const float*)d_in[17];
    const float* b_cp    = (const float*)d_in[18];
    const float* lnpo_g  = (const float*)d_in[19];
    const float* lnpo_b  = (const float*)d_in[20];
    const float* proj    = (const float*)d_in[21];
    float* out = (float*)d_out;
    (void)in_sizes; (void)n_in; (void)out_size;

    char*  wsp = (char*)d_ws;
    size_t off = 0;
    auto alloc = [&](size_t bytes) -> char* {
        char* p = wsp + off;
        off += (bytes + 255) & ~(size_t)255;
        return p;
    };
    u16*   wc_b   = (u16*)alloc((size_t)DD * DD * 2);
    u16*   win_b  = (u16*)alloc((size_t)NL * TD * DD * 2);
    u16*   wout_b = (u16*)alloc((size_t)NL * DD * DD * 2);
    u16*   wfc_b  = (u16*)alloc((size_t)NL * MLPD * DD * 2);
    u16*   wcp_b  = (u16*)alloc((size_t)NL * DD * MLPD * 2);
    u16*   proj_b = (u16*)alloc((size_t)DD * 512 * 2);
    float* X      = (float*)alloc((size_t)TOK * DD * 4);
    u16*   XN     = (u16*)alloc((size_t)TOK * DD * 2);       // ln out; reused as attn-out (o)
    u16*   QKV    = (u16*)alloc((size_t)TOK * TD * 2);
    float* S      = (float*)alloc((size_t)BB * HH * NTOK * SLD * 4);  // scores; reused as patch fp32
    u16*   Hbuf   = (u16*)alloc((size_t)TOK * MLPD * 2);     // MLP hidden; reused as im2col bf16
    u16*   XCLS   = (u16*)alloc((size_t)BB * DD * 2);
    if (off > ws_size) return;   // scratch too small: do nothing (deterministic)

    const dim3 blk256(256);
    auto cvt = [&](const float* src, u16* dst, long n) {
        f32_to_bf16k<<<dim3((unsigned)((n + 255) / 256)), blk256, 0, stream>>>(src, dst, n);
    };
    auto gemm_nt = [&](const u16* A, long lda, const u16* Bm, long ldb,
                       const float* bias, float* Cf, u16* Cb, long ldc, const float* Res,
                       int M, int N, int K, int act, int Z, int zdiv,
                       long sA0, long sA1, long sB0, long sB1, long sC0, long sC1) {
        dim3 grid((unsigned)((N + 127) / 128), (unsigned)((M + 127) / 128), (unsigned)Z);
        gemm_nt_tdm<<<grid, dim3(32, 8), 0, stream>>>(
            A, lda, Bm, ldb, bias, Cf, Cb, ldc, Res,
            M, N, K, act, zdiv, sA0, sA1, sB0, sB1, sC0, sC1);
    };
    auto gemm_v = [&](const u16* A, long lda, const u16* Bm, long ldb, int Kvalid,
                      const float* bias, float* Cf, u16* Cb, long ldc,
                      int M, int N, int K, int Z, int zdiv,
                      long sA0, long sA1, long sB0, long sB1, long sC0, long sC1) {
        dim3 grid((unsigned)((N + 63) / 64), (unsigned)((M + 127) / 128), (unsigned)Z);
        gemm_kn<<<grid, dim3(32, 8), 0, stream>>>(
            A, lda, Bm, ldb, Kvalid, bias, Cf, Cb, ldc,
            M, N, K, zdiv, sA0, sA1, sB0, sB1, sC0, sC1);
    };

    // -------- one-time weight conversion to bf16 --------
    cvt(conv_w, wc_b,   (long)DD * DD);
    cvt(w_in,   win_b,  (long)NL * TD * DD);
    cvt(w_out,  wout_b, (long)NL * DD * DD);
    cvt(w_fc,   wfc_b,  (long)NL * MLPD * DD);
    cvt(w_cp,   wcp_b,  (long)NL * DD * MLPD);
    cvt(proj,   proj_b, (long)DD * 512);

    // -------- patch embedding: im2col + GEMM, then cls/pos assemble --------
    u16*   Pim    = Hbuf;   // [12544, 768] bf16
    float* patchC = S;      // [12544, 768] fp32 (scores region, free right now)
    im2col_patches<<<dim3((unsigned)(((long)PTOK * DD + 255) / 256)), blk256, 0, stream>>>(x_inp, Pim);
    gemm_nt(Pim, DD, wc_b, DD, nullptr, patchC, nullptr, DD, nullptr,
            PTOK, DD, DD, 0, 1, 1, 0, 0, 0, 0, 0, 0);
    assemble_embed<<<dim3((unsigned)(((long)TOK * DD + 255) / 256)), blk256, 0, stream>>>(patchC, cls, pos, X);
    layernorm768<<<dim3(TOK), blk256, 0, stream>>>(X, DD, lnpre_g, lnpre_b, X, nullptr, DD);  // ln_pre in-place

    // -------- 12 transformer layers --------
    for (int l = 0; l < NL; ++l) {
        const u16* wi  = win_b  + (size_t)l * TD * DD;
        const u16* wo  = wout_b + (size_t)l * DD * DD;
        const u16* wf  = wfc_b  + (size_t)l * MLPD * DD;
        const u16* wcl = wcp_b  + (size_t)l * DD * MLPD;

        layernorm768<<<dim3(TOK), blk256, 0, stream>>>(X, DD, ln1_g + (long)l * DD, ln1_b + (long)l * DD,
                                                       nullptr, XN, DD);
        // qkv = xn @ w_in^T + b_in       [12608, 2304] bf16
        gemm_nt(XN, DD, wi, DD, b_in + (long)l * TD, nullptr, QKV, TD, nullptr,
                TOK, TD, DD, 0, 1, 1, 0, 0, 0, 0, 0, 0);
        // scores[b,h] = q @ k^T          batched over z = b*12 + h
        gemm_nt(QKV, TD, QKV + DD, TD, nullptr, S, nullptr, SLD, nullptr,
                NTOK, NTOK, 64, 0, BB * HH, HH,
                (long)NTOK * TD, 64, (long)NTOK * TD, 64,
                (long)HH * NTOK * SLD, (long)NTOK * SLD);
        softmax_mask<<<dim3(BB * HH * NTOK), blk256, 0, stream>>>(S, maskp);
        // o[b,h] = P @ v  (P bf16 zero-padded K=224, embedded in score rows) -> XN as o
        gemm_v((const u16*)S + SLD, PLD, QKV + 2 * DD, TD, NTOK, nullptr, nullptr, XN, DD,
               NTOK, 64, SLD, BB * HH, HH,
               (long)HH * NTOK * PLD, (long)NTOK * PLD,
               (long)NTOK * TD, 64,
               (long)NTOK * DD, 64);
        // x += o @ w_out^T + b_out
        gemm_nt(XN, DD, wo, DD, b_out + (long)l * DD, X, nullptr, DD, X,
                TOK, DD, DD, 0, 1, 1, 0, 0, 0, 0, 0, 0);
        // MLP
        layernorm768<<<dim3(TOK), blk256, 0, stream>>>(X, DD, ln2_g + (long)l * DD, ln2_b + (long)l * DD,
                                                       nullptr, XN, DD);
        gemm_nt(XN, DD, wf, DD, b_fc + (long)l * MLPD, nullptr, Hbuf, MLPD, nullptr,
                TOK, MLPD, DD, 1 /*QuickGELU*/, 1, 1, 0, 0, 0, 0, 0, 0);
        gemm_nt(Hbuf, MLPD, wcl, MLPD, b_cp + (long)l * DD, X, nullptr, DD, X,
                TOK, DD, MLPD, 0, 1, 1, 0, 0, 0, 0, 0, 0);
    }

    // -------- ln_post on cls tokens + projection --------
    layernorm768<<<dim3(BB), blk256, 0, stream>>>(X, (long)NTOK * DD, lnpo_g, lnpo_b, nullptr, XCLS, DD);
    gemm_v(XCLS, DD, proj_b, 512, DD, nullptr, out, nullptr, 512,
           BB, 512, DD, 1, 1, 0, 0, 0, 0, 0, 0);
}
